// GAT_49168785605294
// MI455X (gfx1250) — matrix operand order repaired and checked by reading, not compiled, not run on hardware
//
#include <hip/hip_runtime.h>
#include <hip/hip_bf16.h>

// ---------------------------------------------------------------------------
// Sizes from the reference
// ---------------------------------------------------------------------------
#define NODES   128
#define LATENT  32
#define GRU     4096          // NODES*LATENT
#define G3      12288         // 3*GRU
#define HIDDEN  1024
#define T_IN    32
#define T_OUT   64

typedef __attribute__((ext_vector_type(16))) __bf16 v16bf;
typedef __attribute__((ext_vector_type(8)))  __bf16 v8bf;
typedef __attribute__((ext_vector_type(8)))  float  v8f;

// ---------------------------------------------------------------------------
// Workspace layout (bytes). Total ~201.5 MB (dominated by bf16 weight copies).
// ---------------------------------------------------------------------------
static const size_t OFF_WIH = 0;                               // 12288*4096 bf16 (transposed)
static const size_t OFF_WHH = OFF_WIH + (size_t)G3 * GRU * 2;  // 12288*4096 bf16 (transposed)
static const size_t OFF_GI  = OFF_WHH + (size_t)G3 * GRU * 2;  // 12288 f32
static const size_t OFF_GH  = OFF_GI  + (size_t)G3 * 4;        // 12288 f32
static const size_t OFF_XB  = OFF_GH  + (size_t)G3 * 4;        // 4096 bf16
static const size_t OFF_HB  = OFF_XB  + (size_t)GRU * 2;       // 4096 bf16
static const size_t OFF_HID = OFF_HB  + (size_t)GRU * 2;       // 4096 f32
static const size_t OFF_H0A = OFF_HID + (size_t)GRU * 4;       // 1024 f32   } contiguous ->
static const size_t OFF_TAB = OFF_H0A + (size_t)HIDDEN * 4;    // 1024 f32   } concat([h0,table])
static const size_t OFF_H0B = OFF_TAB + (size_t)HIDDEN * 4;    // 1024 f32
static const size_t OFF_H0C = OFF_H0B + (size_t)HIDDEN * 4;    // 1024 f32
static const size_t OFF_ZB  = OFF_H0C + (size_t)HIDDEN * 4;    // 256 B of zeros (A-pad rows)

// ---------------------------------------------------------------------------
// 1) fp32 -> bf16 conversion with transpose, LDS-tiled.
//    src: W[k][j], k<4096 rows, j<12288 cols (row-major, fan_out contiguous)
//    dst: W_t[j][k]  (k contiguous -> per-lane contiguous WMMA B fragments)
// ---------------------------------------------------------------------------
__global__ void convert_transpose(const float* __restrict__ Wih,
                                  const float* __restrict__ Whh,
                                  __bf16* __restrict__ outIh,
                                  __bf16* __restrict__ outHh) {
  const float* src = blockIdx.z ? Whh : Wih;
  __bf16*      dst = blockIdx.z ? outHh : outIh;
  __shared__ float tile[32][33];
  const int j0 = blockIdx.x * 32;     // column in src
  const int k0 = blockIdx.y * 32;     // row in src
  const int tx = threadIdx.x, ty = threadIdx.y;   // 32 x 8
  #pragma unroll
  for (int i = 0; i < 32; i += 8)
    tile[ty + i][tx] = src[(size_t)(k0 + ty + i) * G3 + j0 + tx];
  __syncthreads();
  #pragma unroll
  for (int i = 0; i < 32; i += 8)
    dst[(size_t)(j0 + ty + i) * GRU + k0 + tx] = (__bf16)tile[tx][ty + i];
}

// ---------------------------------------------------------------------------
// 2) init: hidden = 0, hb = 0, x(t=0) = h[0]*W1 + b1 (bf16), zero-pad block
// ---------------------------------------------------------------------------
__global__ void gru_init(const float* __restrict__ h,
                         const float* __restrict__ W1,
                         const float* __restrict__ b1,
                         float* __restrict__ hidden,
                         __bf16* __restrict__ hb,
                         __bf16* __restrict__ xb,
                         float* __restrict__ zb) {
  const int j = blockIdx.x * 256 + threadIdx.x;   // 0..4095
  hidden[j] = 0.0f;
  hb[j]     = (__bf16)0.0f;
  const int n = j >> 5, c = j & 31;
  xb[j] = (__bf16)(h[n] * W1[c] + b1[c]);         // h[0, n, 0] == h[n]
  if (j < 64) zb[j] = 0.0f;                       // 256 B zero block for A padding
}

// ---------------------------------------------------------------------------
// 3) The dominant kernel: dual GEMV via WMMA bf16.
//    out[jb+n] = sum_k vec[k] * W[k][jb+n]   (weights stored transposed, bf16)
//    One wave per 32-column tile (two 16x16 WMMA accumulators -> 2 independent
//    accumulation chains for ILP; A fragment shared between both).
//    A tile = 16x32 with the vector in row M=0: contiguous K order (as B),
//    lane 0 holds K {0..15}, lane 16 holds K {16..31}; all other
//    lanes read a resident 256B zero block via a per-lane pointer select
//    (no divergent branch, no per-iteration re-zeroing).
//    grid.y selects (x,Wih)->gi vs (hidden,Whh)->gh.
// ---------------------------------------------------------------------------
__global__ void gru_gemv_wmma(const __bf16* __restrict__ WT0,
                              const __bf16* __restrict__ WT1,
                              const __bf16* __restrict__ xb,
                              const __bf16* __restrict__ hb,
                              float* __restrict__ gi,
                              float* __restrict__ gh,
                              const __bf16* __restrict__ zb) {
  const int mat = blockIdx.y;
  const __bf16* WT  = mat ? WT1 : WT0;
  const __bf16* vec = mat ? hb  : xb;
  float*        out = mat ? gh  : gi;

  const int jb   = blockIdx.x * 32;      // 32 output columns per wave
  const int lane = threadIdx.x;          // 0..31
  const int m    = lane & 15;            // row (A) / column (B)
  const int hi   = lane >> 4;            // K-half select

  // B: lane holds column jb+m (tile 0) / jb+16+m (tile 1),
  //    16 consecutive K values (K-half selected by lane group).
  const __bf16* brow0 = WT + (size_t)(jb + m)      * GRU + hi * 16;
  const __bf16* brow1 = WT + (size_t)(jb + 16 + m) * GRU + hi * 16;

  // A: row M=0 carries the vector; other rows read the zero block.
  const __bf16* aptr  = (m == 0) ? (vec + (hi << 4)) : zb;
  const int     astep = (m == 0) ? 32 : 0;

  union AU { v16bf v; v8bf h[2]; };

  v8f acc0 = {};
  v8f acc1 = {};
  for (int kc = 0; kc < GRU; kc += 32) {
    AU a;
    a.h[0] = *(const v8bf*)(aptr);
    a.h[1] = *(const v8bf*)(aptr + 8);
    aptr += astep;
    v16bf b0 = *(const v16bf*)(brow0 + kc);
    v16bf b1 = *(const v16bf*)(brow1 + kc);
    __builtin_prefetch(brow0 + kc + 256, 0, 1);   // global_prefetch_b8, 512B ahead
    __builtin_prefetch(brow1 + kc + 256, 0, 1);
    acc0 = __builtin_amdgcn_wmma_f32_16x16x32_bf16(
        false, a.v, false, b0, (short)0, acc0, false, false);
    acc1 = __builtin_amdgcn_wmma_f32_16x16x32_bf16(
        false, a.v, false, b1, (short)0, acc1, false, false);
  }
  // D row M=0 lives in VGPR0 of lanes 0..15 (N = lane)
  if (lane < 16) {
    out[jb + lane]      = acc0[0];
    out[jb + 16 + lane] = acc1[0];
  }
}

// ---------------------------------------------------------------------------
// 4) Fused per-step tail: GRU gating -> hnew (LDS), hidden' = out@W2+b2,
//    temp' = out@Wh2i+bh2i, x(t+1) = temp'*W1+b1 (or from h while t+1<32).
//    Block = 256 threads = 8 nodes; j == blockIdx*256 + tid.
// ---------------------------------------------------------------------------
__global__ void gru_update(const float* __restrict__ gi,
                           const float* __restrict__ gh,
                           const float* __restrict__ bih,
                           const float* __restrict__ bhh,
                           float* __restrict__ hidden,
                           __bf16* __restrict__ hb,
                           __bf16* __restrict__ xb,
                           const float* __restrict__ W2,
                           const float* __restrict__ b2,
                           const float* __restrict__ Wh2i,
                           const float* __restrict__ bh2i,
                           const float* __restrict__ W1,
                           const float* __restrict__ b1,
                           const float* __restrict__ h,
                           int tnext) {
  __shared__ float hn_s[256];
  __shared__ float temp_s[8];
  const int tid = threadIdx.x;
  const int j   = blockIdx.x * 256 + tid;   // 0..4095

  // ---- phase 1: gating ----
  {
    float ir = gi[j]        + bih[j];
    float hr = gh[j]        + bhh[j];
    float iz = gi[GRU + j]  + bih[GRU + j];
    float hz = gh[GRU + j]  + bhh[GRU + j];
    float in_= gi[2*GRU + j]+ bih[2*GRU + j];
    float hn = gh[2*GRU + j]+ bhh[2*GRU + j];
    float r = 1.0f / (1.0f + expf(-(ir + hr)));
    float z = 1.0f / (1.0f + expf(-(iz + hz)));
    float n = tanhf(in_ + r * hn);
    hn_s[tid] = (1.0f - z) * n + z * hidden[j];
  }
  __syncthreads();

  const int nl = tid >> 5;                 // local node 0..7
  const int c  = tid & 31;                 // latent channel
  const int n  = blockIdx.x * 8 + nl;      // global node

  // ---- phase 2: hidden' = out@W2 + b2 ; temp' = out@Wh2i + bh2i ----
  {
    float acc = b2[c];
    #pragma unroll 8
    for (int l = 0; l < 32; ++l) acc += hn_s[nl * 32 + l] * W2[l * 32 + c];
    hidden[n * 32 + c] = acc;
    hb[n * 32 + c]     = (__bf16)acc;
    if (c == 0) {
      float t2 = bh2i[0];
      #pragma unroll 8
      for (int l = 0; l < 32; ++l) t2 += hn_s[nl * 32 + l] * Wh2i[l];
      temp_s[nl] = t2;
    }
  }
  __syncthreads();

  // ---- phase 3: x for next step ----
  {
    float tv = (tnext < T_IN) ? h[tnext * NODES + n] : temp_s[nl];
    xb[n * 32 + c] = (__bf16)(tv * W1[c] + b1[c]);
  }
}

// ---------------------------------------------------------------------------
// 5) Epilogue GEMV (fp32, runs once; coalesced across output columns).
//    out[k] = bias[k] + sum_j v[j] * W[j*ncols + k]
// ---------------------------------------------------------------------------
__global__ void gemv_f32(const float* __restrict__ v, int nrows,
                         const float* __restrict__ W,
                         const float* __restrict__ bias,
                         float* __restrict__ out, int ncols) {
  const int k = blockIdx.x * 256 + threadIdx.x;
  if (k >= ncols) return;
  float acc = bias[k];
  for (int j = 0; j < nrows; ++j)
    acc = fmaf(v[j], W[(size_t)j * ncols + k], acc);
  out[k] = acc;
}

// ---------------------------------------------------------------------------
// Host-side orchestration
// ---------------------------------------------------------------------------
extern "C" void kernel_launch(void* const* d_in, const int* in_sizes, int n_in,
                              void* d_out, int out_size, void* d_ws, size_t ws_size,
                              hipStream_t stream) {
  (void)in_sizes; (void)n_in; (void)out_size; (void)ws_size;
  const float* h    = (const float*)d_in[0];
  // d_in[1] = target (unused)
  const float* W1   = (const float*)d_in[2];
  const float* b1   = (const float*)d_in[3];
  const float* W2   = (const float*)d_in[4];
  const float* b2   = (const float*)d_in[5];
  const float* Wt   = (const float*)d_in[6];
  const float* bt   = (const float*)d_in[7];
  const float* Wh2i = (const float*)d_in[8];
  const float* bh2i = (const float*)d_in[9];
  const float* Wih  = (const float*)d_in[10];
  const float* Whh  = (const float*)d_in[11];
  const float* bih  = (const float*)d_in[12];
  const float* bhh  = (const float*)d_in[13];
  const float* Wh1  = (const float*)d_in[14];
  const float* bh1  = (const float*)d_in[15];
  const float* Wh2a = (const float*)d_in[16];
  const float* bh2a = (const float*)d_in[17];
  const float* Wh3  = (const float*)d_in[18];
  const float* bh3  = (const float*)d_in[19];
  const float* Whh2 = (const float*)d_in[20];
  const float* bhh2 = (const float*)d_in[21];

  char* ws = (char*)d_ws;
  __bf16* WT_ih  = (__bf16*)(ws + OFF_WIH);
  __bf16* WT_hh  = (__bf16*)(ws + OFF_WHH);
  float*  gi     = (float*) (ws + OFF_GI);
  float*  gh     = (float*) (ws + OFF_GH);
  __bf16* xb     = (__bf16*)(ws + OFF_XB);
  __bf16* hb     = (__bf16*)(ws + OFF_HB);
  float*  hidden = (float*) (ws + OFF_HID);
  float*  h0a    = (float*) (ws + OFF_H0A);   // [h0a, tab] contiguous (2048 f32)
  float*  tab    = (float*) (ws + OFF_TAB);
  float*  h0b    = (float*) (ws + OFF_H0B);
  float*  h0c    = (float*) (ws + OFF_H0C);
  float*  zbf    = (float*) (ws + OFF_ZB);
  const __bf16* zb = (const __bf16*)(ws + OFF_ZB);

  // Weight conversion + transpose (once per call): 576 MB traffic ~= 25 us,
  // buys 192 MB/step instead of 384 MB/step for 64 steps (bf16 set ~= L2 size).
  convert_transpose<<<dim3(G3 / 32, GRU / 32, 2), dim3(32, 8), 0, stream>>>(
      Wih, Whh, WT_ih, WT_hh);

  gru_init<<<GRU / 256, 256, 0, stream>>>(h, W1, b1, hidden, hb, xb, zbf);

  for (int t = 0; t < T_OUT; ++t) {
    gru_gemv_wmma<<<dim3(G3 / 32, 2), 32, 0, stream>>>(WT_ih, WT_hh, xb, hb,
                                                       gi, gh, zb);
    gru_update<<<GRU / 256, 256, 0, stream>>>(gi, gh, bih, bhh, hidden, hb, xb,
                                              W2, b2, Wh2i, bh2i, W1, b1, h, t + 1);
  }

  // Epilogue MLP (batch 1, fp32, ~37 MB total traffic)
  gemv_f32<<<HIDDEN / 256, 256, 0, stream>>>(h + 14, 7,        Wt,   bt,   tab,  HIDDEN);
  gemv_f32<<<HIDDEN / 256, 256, 0, stream>>>(hidden, GRU,      Wh1,  bh1,  h0a,  HIDDEN);
  gemv_f32<<<HIDDEN / 256, 256, 0, stream>>>(h0a,    2*HIDDEN, Wh2a, bh2a, h0b,  HIDDEN);
  gemv_f32<<<HIDDEN / 256, 256, 0, stream>>>(h0b,    HIDDEN,   Wh3,  bh3,  h0c,  HIDDEN);
  gemv_f32<<<2 * HIDDEN / 256, 256, 0, stream>>>(h0c, HIDDEN,  Whh2, bhh2,
                                                 (float*)d_out, 2 * HIDDEN);
}
